// adjLoss_33535104647185
// MI455X (gfx1250) — 8-bit matrix operand order repaired and checked by reading, compiled, not run on hardware
//
#include <hip/hip_runtime.h>


#define NUM_CLASSES 80
#define BATCH       1048576
#define CHUNK       256
#define NCHUNK      (BATCH / CHUNK)
#define LDS_STRIDE  272
#define GEMM_BLOCKS 512
#define EPS_F       1e-7f

typedef int v8i __attribute__((ext_vector_type(8)));

__global__ void zero_cooc(int* __restrict__ cooc) {
    int i = blockIdx.x * blockDim.x + threadIdx.x;
    if (i < NUM_CLASSES * NUM_CLASSES) cooc[i] = 0;
}

__device__ __forceinline__ unsigned lbit(unsigned u) { return (u >> 23) & 1u; }

__global__ __launch_bounds__(256) void cooc_gemm(const float* __restrict__ label,
                                                 int* __restrict__ cooc) {
    __shared__ __align__(16) unsigned char tileT[NUM_CLASSES * LDS_STRIDE];

    const int tid  = threadIdx.x;
    const int lane = tid & 31;
    const int wave = tid >> 5;
    const int M    = lane & 15;
    const int half = lane >> 4;

    int tIdx[4], ibs[4], jbs[4];
    #pragma unroll
    for (int s = 0; s < 4; ++s) {
        int t = wave + s * 8;
        tIdx[s] = t;
        if (t > 24) t = 24;
        ibs[s] = (t / 5) * 16;
        jbs[s] = (t % 5) * 16;
    }

    v8i acc[4] = {};

    for (int chunk = blockIdx.x; chunk < NCHUNK; chunk += gridDim.x) {
        const size_t rowBase = (size_t)chunk * CHUNK;
        #pragma unroll
        for (int s = 0; s < 5; ++s) {
            const int q   = tid + s * 256;
            const int brG = q / 20;
            const int c4  = q % 20;
            const float* base = label + (rowBase + (size_t)brG * 4) * NUM_CLASSES + c4 * 4;
            const uint4 r0 = *(const uint4*)(base);
            const uint4 r1 = *(const uint4*)(base + NUM_CLASSES);
            const uint4 r2 = *(const uint4*)(base + 2 * NUM_CLASSES);
            const uint4 r3 = *(const uint4*)(base + 3 * NUM_CLASSES);
            const unsigned p0 = lbit(r0.x) | (lbit(r1.x) << 8) | (lbit(r2.x) << 16) | (lbit(r3.x) << 24);
            const unsigned p1 = lbit(r0.y) | (lbit(r1.y) << 8) | (lbit(r2.y) << 16) | (lbit(r3.y) << 24);
            const unsigned p2 = lbit(r0.z) | (lbit(r1.z) << 8) | (lbit(r2.z) << 16) | (lbit(r3.z) << 24);
            const unsigned p3 = lbit(r0.w) | (lbit(r1.w) << 8) | (lbit(r2.w) << 16) | (lbit(r3.w) << 24);
            unsigned char* dst = tileT + (c4 * 4) * LDS_STRIDE + brG * 4;
            *(unsigned*)(dst)                  = p0;
            *(unsigned*)(dst + 1 * LDS_STRIDE) = p1;
            *(unsigned*)(dst + 2 * LDS_STRIDE) = p2;
            *(unsigned*)(dst + 3 * LDS_STRIDE) = p3;
        }
        __syncthreads();

        #pragma unroll
        for (int s = 0; s < 4; ++s) {
            const int ib = ibs[s];
            const int jb = jbs[s];
            #pragma unroll
            for (int ks = 0; ks < 4; ++ks) {
                const int kb = ks * 64;
                const unsigned char* arow = tileT + (ib + M) * LDS_STRIDE + kb + half * 16;
                const int2 a01 = *(const int2*)(arow);
                const int2 a23 = *(const int2*)(arow + 8);
                const int2 a45 = *(const int2*)(arow + 32);
                const int2 a67 = *(const int2*)(arow + 40);
                const unsigned char* brow = tileT + (jb + M) * LDS_STRIDE + kb + half * 16;
                const int4 b03 = *(const int4*)(brow);
                const int4 b47 = *(const int4*)(brow + 32);

                v8i av, bv;
                av[0] = a01.x; av[1] = a01.y; av[2] = a23.x; av[3] = a23.y;
                av[4] = a45.x; av[5] = a45.y; av[6] = a67.x; av[7] = a67.y;
                bv[0] = b03.x; bv[1] = b03.y; bv[2] = b03.z; bv[3] = b03.w;
                bv[4] = b47.x; bv[5] = b47.y; bv[6] = b47.z; bv[7] = b47.w;

                acc[s] = __builtin_amdgcn_wmma_i32_16x16x64_iu8(
                     false, av,  false, bv,
                    acc[s],  false,  false);
            }
        }
        __syncthreads();
    }

    #pragma unroll
    for (int s = 0; s < 4; ++s) {
        if (tIdx[s] < 25) {
            const int ib = ibs[s];
            const int jb = jbs[s];
            #pragma unroll
            for (int v = 0; v < 8; ++v) {
                const int mi = ib + v + half * 8;
                const int nj = jb + M;
                atomicAdd(&cooc[mi * NUM_CLASSES + nj], acc[s][v]);
            }
        }
    }
}

__global__ __launch_bounds__(256) void finalize(const float* __restrict__ pre_adj,
                                                const int* __restrict__ cooc,
                                                float* __restrict__ out) {
    __shared__ float red[256];
    const int tid = threadIdx.x;
    float local = 0.0f;
    for (int idx = tid; idx < NUM_CLASSES * NUM_CLASSES; idx += 256) {
        const int i = idx / NUM_CLASSES;
        const int j = idx % NUM_CLASSES;
        float target;
        if (i == j) {
            target = 1.0f;
        } else {
            const float cnt = (float)cooc[i * NUM_CLASSES + i];
            target = (float)cooc[idx] / (cnt + EPS_F);
        }
        float r = fabsf(pre_adj[idx] - target);
        local += (r < 1.0f) ? (r * r) : (r - 0.5f);
    }
    red[tid] = local;
    __syncthreads();
    #pragma unroll
    for (int s = 128; s > 0; s >>= 1) {
        if (tid < s) red[tid] += red[tid + s];
        __syncthreads();
    }
    if (tid == 0) out[0] = red[0] * (1.0f / (NUM_CLASSES * NUM_CLASSES));
}

extern "C" void kernel_launch(void* const* d_in, const int* in_sizes, int n_in,
                              void* d_out, int out_size, void* d_ws, size_t ws_size,
                              hipStream_t stream) {
    const float* pre_adj = (const float*)d_in[0];
    const float* label   = (const float*)d_in[1];
    float* out = (float*)d_out;
    int* cooc = (int*)d_ws;

    zero_cooc<<<(NUM_CLASSES * NUM_CLASSES + 255) / 256, 256, 0, stream>>>(cooc);
    cooc_gemm<<<GEMM_BLOCKS, 256, 0, stream>>>(label, cooc);
    finalize<<<1, 256, 0, stream>>>(pre_adj, cooc, out);
}
